// PaTS_GNN_38019050504277
// MI455X (gfx1250) — compile-verified, matrix operand order checked by reading, not run on hardware
//
#include <hip/hip_runtime.h>
#include <hip/hip_bf16.h>
#include <stdint.h>

// ---------------------------------------------------------------------------
// Problem constants (from reference)
// ---------------------------------------------------------------------------
#define B_    64
#define T_    32
#define NPER_ 50
#define N_S   (B_ * T_ * NPER_)   // 102400 state nodes
#define N_G   (B_ * NPER_)        // 3200 goal nodes
#define R_    3
#define IN_F  32
#define D_    64
#define H_    128
#define OUT_  256
#define E_S   819200
#define E_G   25600
#define NSEQ  (B_ * (T_ - 1))     // 1984 sequence rows

typedef __attribute__((ext_vector_type(16))) __bf16 v16bf;
typedef __attribute__((ext_vector_type(8)))  __bf16 v8bf;
typedef __attribute__((ext_vector_type(8)))  float  v8f;
typedef unsigned int u32;

// ---------------------------------------------------------------------------
// Order-preserving float<->uint mapping for atomic max on signed floats
// ---------------------------------------------------------------------------
__device__ __forceinline__ u32 fmap(float f) {
  u32 u = __float_as_uint(f);
  return (u & 0x80000000u) ? ~u : (u | 0x80000000u);
}
// mapped(-inf) = ~0xFF800000 = 0x007FFFFF
#define MAPPED_NEG_INF 0x007FFFFFu

__device__ __forceinline__ float sigmoidf_(float x) { return 1.0f / (1.0f + expf(-x)); }

// ---------------------------------------------------------------------------
// Elementwise prep kernels
// ---------------------------------------------------------------------------
__global__ void k_f32_to_bf16(const float* __restrict__ s, __bf16* __restrict__ d, size_t n) {
  size_t t = (size_t)blockIdx.x * blockDim.x + threadIdx.x;
  if (t < n) d[t] = (__bf16)s[t];
}

// W is KxN row-major f32; write Wt = W^T as NxK row-major bf16
__global__ void k_transpose_to_bf16(const float* __restrict__ W, __bf16* __restrict__ Wt,
                                    int K, int N) {
  int t = blockIdx.x * blockDim.x + threadIdx.x;
  if (t >= K * N) return;
  int k = t / N, n = t - k * N;
  Wt[(size_t)n * K + k] = (__bf16)W[t];
}

__global__ void k_fill_u32(u32* __restrict__ p, u32 v, size_t n) {
  size_t t = (size_t)blockIdx.x * blockDim.x + threadIdx.x;
  if (t < n) p[t] = v;
}

__global__ void k_add_bias(const float* __restrict__ a, const float* __restrict__ b,
                           float* __restrict__ o, int n) {
  int t = blockIdx.x * blockDim.x + threadIdx.x;
  if (t < n) o[t] = a[t] + b[t];
}

// ---------------------------------------------------------------------------
// bf16 WMMA GEMM:  C(MxN,f32) = A(MxK,bf16) @ Bt^T + bias, opt ReLU, optional
// bf16 mirror of C.  Bt is B^T stored NxK row-major.
//
// Block = 256 threads (8 wave32), 64x64 output tile, K-step 32, double-buffered
// LDS.  Waves arranged 4(M) x 2(N); each wave owns a 16x32 sub-tile: one A
// fragment reused across two B fragments -> 2 v_wmma per K-step per wave.
//
// Staging: each thread owns one 16B segment of each 64x32 bf16 tile ->
// 1 global_load_b128 + 1 ds_store_b128 per operand per K-step, fully coalesced.
// LDS row stride 40 bf16 (80B): 16B-aligned rows; 20-dword stride makes the
// per-lane b128 fragment reads bank-conflict-free.  Fragments are two aligned
// b128 LDS loads (k = half*8..+7 and 16+half*8..+7 are contiguous runs).
// M, N multiples of 64; K multiple of 32 (holds for every call here).
// ---------------------------------------------------------------------------
__global__ __launch_bounds__(256) void k_gemm_bf16(
    const __bf16* __restrict__ A, const __bf16* __restrict__ Bt,
    const float* __restrict__ bias, float* __restrict__ C,
    __bf16* __restrict__ Cbf, int M, int N, int K, int relu) {
  __shared__ __attribute__((aligned(16))) __bf16 lA[2][64][40];
  __shared__ __attribute__((aligned(16))) __bf16 lB[2][64][40];

  const int tid  = threadIdx.x;
  const int lane = tid & 31;
  const int wid  = tid >> 5;
  const int wm   = (wid & 3) << 4;     // wave M offset in tile: 0,16,32,48
  const int wn   = (wid >> 2) << 5;    // wave N offset in tile: 0,32
  const int m0   = blockIdx.y << 6;
  const int n0   = blockIdx.x << 6;

  // staging assignment: thread -> (row, 16B segment) of the 64x32 bf16 tile
  const int lrow = tid >> 2;           // 0..63
  const int lseg = tid & 3;            // segment of 8 bf16 (16B)
  const __bf16* gA = A  + (size_t)(m0 + lrow) * K + lseg * 8;
  const __bf16* gB = Bt + (size_t)(n0 + lrow) * K + lseg * 8;

  const int half = lane >> 4;          // K-half of the fragment layout
  const int frow = lane & 15;          // fragment row (A: M index, B: N index)

  v8f acc0 = {0.f, 0.f, 0.f, 0.f, 0.f, 0.f, 0.f, 0.f};
  v8f acc1 = {0.f, 0.f, 0.f, 0.f, 0.f, 0.f, 0.f, 0.f};

  // prologue: stage K-tile 0 into buffer 0
  *(uint4*)((char*)&lA[0][lrow][0] + 16 * lseg) = *(const uint4*)gA;
  *(uint4*)((char*)&lB[0][lrow][0] + 16 * lseg) = *(const uint4*)gB;
  gA += 32;
  gB += 32;
  __syncthreads();

  const int nk = K >> 5;
  int cur = 0;
  for (int kt = 0; kt < nk; ++kt) {
    if (kt + 1 < nk) {                 // stage next tile into the alternate buffer
      __builtin_prefetch(gA + 32);     // -> global_prefetch_b8 (tile after next)
      __builtin_prefetch(gB + 32);
      *(uint4*)((char*)&lA[cur ^ 1][lrow][0] + 16 * lseg) = *(const uint4*)gA;
      *(uint4*)((char*)&lB[cur ^ 1][lrow][0] + 16 * lseg) = *(const uint4*)gB;
      gA += 32;
      gB += 32;
    }
    // fragments per CDNA5 16-bit A/B VGPR layout (ISA 7.12.2): two b128 loads
    v8bf alo = *(const v8bf*)&lA[cur][wm + frow][half * 8];
    v8bf ahi = *(const v8bf*)&lA[cur][wm + frow][16 + half * 8];
    v8bf b0l = *(const v8bf*)&lB[cur][wn + frow][half * 8];
    v8bf b0h = *(const v8bf*)&lB[cur][wn + frow][16 + half * 8];
    v8bf b1l = *(const v8bf*)&lB[cur][wn + 16 + frow][half * 8];
    v8bf b1h = *(const v8bf*)&lB[cur][wn + 16 + frow][16 + half * 8];
    v16bf af  = __builtin_shufflevector(alo, ahi, 0, 1, 2, 3, 4, 5, 6, 7,
                                        8, 9, 10, 11, 12, 13, 14, 15);
    v16bf bf0 = __builtin_shufflevector(b0l, b0h, 0, 1, 2, 3, 4, 5, 6, 7,
                                        8, 9, 10, 11, 12, 13, 14, 15);
    v16bf bf1 = __builtin_shufflevector(b1l, b1h, 0, 1, 2, 3, 4, 5, 6, 7,
                                        8, 9, 10, 11, 12, 13, 14, 15);
    acc0 = __builtin_amdgcn_wmma_f32_16x16x32_bf16(
        false, af, false, bf0, (short)0, acc0, false, false);
    acc1 = __builtin_amdgcn_wmma_f32_16x16x32_bf16(
        false, af, false, bf1, (short)0, acc1, false, false);
    __syncthreads();
    cur ^= 1;
  }

  // epilogue: C VGPR g holds (M = g + 8*half, N = lane&15)
  const int nc0 = n0 + wn + frow;
  const int nc1 = nc0 + 16;
  const float b0 = bias ? bias[nc0] : 0.0f;
  const float b1 = bias ? bias[nc1] : 0.0f;
#pragma unroll
  for (int g = 0; g < 8; ++g) {
    int m = m0 + wm + g + (half << 3);
    float v0 = acc0[g] + b0;
    float v1 = acc1[g] + b1;
    if (relu) { v0 = fmaxf(v0, 0.0f); v1 = fmaxf(v1, 0.0f); }
    size_t i0 = (size_t)m * N + nc0;
    size_t i1 = (size_t)m * N + nc1;
    if (C)   { C[i0] = v0;             C[i1] = v1; }
    if (Cbf) { Cbf[i0] = (__bf16)v0;   Cbf[i1] = (__bf16)v1; }
  }
}

// ---------------------------------------------------------------------------
// GNN edge kernels
// ---------------------------------------------------------------------------
__global__ void k_scatter_max(const float* __restrict__ msg, const int* __restrict__ src,
                              const int* __restrict__ dst, u32* __restrict__ agg, int E) {
  int t = blockIdx.x * blockDim.x + threadIdx.x;
  if (t >= E * 16) return;
  int e = t >> 4, j = (t & 15) << 2;   // 4 features per thread
  int s = src[e], d = dst[e];
  float4 v = *(const float4*)(msg + (size_t)s * D_ + j);
  u32* a = agg + (size_t)d * D_ + j;
  atomicMax(a + 0, fmap(v.x));
  atomicMax(a + 1, fmap(v.y));
  atomicMax(a + 2, fmap(v.z));
  atomicMax(a + 3, fmap(v.w));
}

__global__ void k_accum_agg(float* __restrict__ out, const u32* __restrict__ agg, size_t n) {
  size_t t = (size_t)blockIdx.x * blockDim.x + threadIdx.x;
  if (t >= n) return;
  u32 u = agg[t];
  u32 bits = (u & 0x80000000u) ? (u & 0x7FFFFFFFu) : ~u;   // un-map
  if (((bits >> 23) & 0xFFu) != 0xFFu)                      // finite only
    out[t] += __uint_as_float(bits);
}

__global__ void k_relu_store(const float* __restrict__ out, float* __restrict__ h32,
                             __bf16* __restrict__ hbf, size_t n) {
  size_t t = (size_t)blockIdx.x * blockDim.x + threadIdx.x;
  if (t >= n) return;
  float v = fmaxf(out[t], 0.0f);
  h32[t] = v;
  hbf[t] = (__bf16)v;
}

// batch ids are arange//NPER -> contiguous groups of NPER rows
__global__ void k_segment_sum(const float* __restrict__ h, float* __restrict__ emb,
                              int n_graphs) {
  int t = blockIdx.x * blockDim.x + threadIdx.x;
  if (t >= n_graphs * D_) return;
  int g = t >> 6, j = t & 63;
  const float* p = h + (size_t)g * NPER_ * D_ + j;
  float s = 0.f;
#pragma unroll 5
  for (int i = 0; i < NPER_; ++i) s += p[(size_t)i * D_];
  emb[t] = s;
}

// ---------------------------------------------------------------------------
// Sequence assembly: X0[b*31+t] = concat(state_emb[b*32+t], goal_emb[b]) (bf16)
// and tgt[b*31+t] = state_emb[b*32+t+1] (f32, straight to d_out)
// ---------------------------------------------------------------------------
__global__ void k_build_input(const float* __restrict__ emb_s, const float* __restrict__ emb_g,
                              __bf16* __restrict__ X0, float* __restrict__ tgt) {
  int t = blockIdx.x * blockDim.x + threadIdx.x;
  if (t >= NSEQ * D_) return;
  int r = t >> 6, j = t & 63;
  int b = r / (T_ - 1), tt = r - b * (T_ - 1);
  X0[(size_t)r * (2 * D_) + j]      = (__bf16)emb_s[(size_t)(b * T_ + tt) * D_ + j];
  X0[(size_t)r * (2 * D_) + D_ + j] = (__bf16)emb_g[(size_t)b * D_ + j];
  tgt[t] = emb_s[(size_t)(b * T_ + tt + 1) * D_ + j];
}

// ---------------------------------------------------------------------------
// LSTM pointwise: gates(64x512) = h@Whh^T (no bias); xpre holds X@Wih^T+b_ih+b_hh
// ---------------------------------------------------------------------------
__global__ void k_lstm_point(const float* __restrict__ gates, const float* __restrict__ xpre,
                             float* __restrict__ c, __bf16* __restrict__ hcur,
                             __bf16* __restrict__ hs, int t) {
  int idx = blockIdx.x * blockDim.x + threadIdx.x;
  if (idx >= B_ * H_) return;
  int b = idx >> 7, j = idx & 127;
  const float* xr = xpre  + (size_t)(b * (T_ - 1) + t) * (4 * H_);
  const float* gr = gates + (size_t)b * (4 * H_);
  float gi = gr[j]           + xr[j];
  float gf = gr[H_ + j]      + xr[H_ + j];
  float gg = gr[2 * H_ + j]  + xr[2 * H_ + j];
  float go = gr[3 * H_ + j]  + xr[3 * H_ + j];
  float cc = sigmoidf_(gf) * c[idx] + sigmoidf_(gi) * tanhf(gg);
  c[idx] = cc;
  float h = sigmoidf_(go) * tanhf(cc);
  hcur[idx] = (__bf16)h;
  hs[(size_t)(b * (T_ - 1) + t) * H_ + j] = (__bf16)h;
}

// ---------------------------------------------------------------------------
// Host-side orchestration
// ---------------------------------------------------------------------------
// Input layout assumption: d_in[0..5] = x_states, edge_index_states,
// batch_states, x_goal, edge_index_goal, batch_goal (dict insertion order);
// then params flattened jax.tree_util-style (dict keys sorted, lists in order):
//   6:dec_b1 7:dec_b2 8:dec_w1 9:dec_w2 10:emb_b 11:emb_w 12:head_b 13:head_w
//   14+5l: conv_w0, conv_w1, conv_w2, root_b, root_w        (l = 0..3)
//   34+4l: b_hh, b_ih, w_hh, w_ih                           (l = 0..1)
enum {
  IN_X_STATES = 0, IN_EDGE_S, IN_BATCH_S, IN_X_GOAL, IN_EDGE_G, IN_BATCH_G,
  IP_DEC_B1, IP_DEC_B2, IP_DEC_W1, IP_DEC_W2, IP_EMB_B, IP_EMB_W,
  IP_HEAD_B, IP_HEAD_W, IP_LAYER0 = 14, IP_LSTM0 = 34
};

static inline unsigned nb256(size_t n) { return (unsigned)((n + 255) / 256); }

extern "C" void kernel_launch(void* const* d_in, const int* in_sizes, int n_in,
                              void* d_out, int out_size, void* d_ws, size_t ws_size,
                              hipStream_t stream) {
  (void)in_sizes; (void)n_in; (void)out_size; (void)ws_size;

  // ---- workspace bump allocator (256B aligned) ----
  char* wsp = (char*)d_ws;
  size_t off = 0;
  auto alloc = [&](size_t bytes) -> void* {
    off = (off + 255) & ~(size_t)255;
    void* p = wsp + off;
    off += bytes;
    return p;
  };

  // activations (sized for the larger states-encode; goal reuses)
  __bf16* x_bf  = (__bf16*)alloc((size_t)N_S * IN_F * 2);
  __bf16* h_bf  = (__bf16*)alloc((size_t)N_S * D_ * 2);
  float*  h32   = (float*) alloc((size_t)N_S * D_ * 4);
  float*  out32 = (float*) alloc((size_t)N_S * D_ * 4);
  float*  msg32 = (float*) alloc((size_t)N_S * D_ * 4);
  u32*    agg   = (u32*)   alloc((size_t)N_S * D_ * 4);
  float*  emb_s = (float*) alloc((size_t)B_ * T_ * D_ * 4);
  float*  emb_g = (float*) alloc((size_t)B_ * D_ * 4);

  // bf16 weights (all stored as B^T, i.e. NxK)
  __bf16* emb_wt  = (__bf16*)alloc((size_t)D_ * IN_F * 2);
  __bf16* root_wt = (__bf16*)alloc((size_t)4 * D_ * D_ * 2);
  __bf16* conv_wt = (__bf16*)alloc((size_t)12 * D_ * D_ * 2);
  __bf16* head_wt = (__bf16*)alloc((size_t)D_ * H_ * 2);
  __bf16* d1t     = (__bf16*)alloc((size_t)H_ * D_ * 2);
  __bf16* d2t     = (__bf16*)alloc((size_t)OUT_ * H_ * 2);
  __bf16* wih[2], *whh[2];
  float*  bsum[2];
  for (int l = 0; l < 2; ++l) {
    wih[l]  = (__bf16*)alloc((size_t)4 * H_ * H_ * 2);   // (512x128) already NxK
    whh[l]  = (__bf16*)alloc((size_t)4 * H_ * H_ * 2);
    bsum[l] = (float*) alloc((size_t)4 * H_ * 4);
  }

  // sequence buffers
  __bf16* X0bf   = (__bf16*)alloc((size_t)NSEQ * 2 * D_ * 2);
  float*  Xpre   = (float*) alloc((size_t)NSEQ * 4 * H_ * 4);
  float*  gates  = (float*) alloc((size_t)B_ * 4 * H_ * 4);
  __bf16* hcur   = (__bf16*)alloc((size_t)B_ * H_ * 2);
  float*  cbuf   = (float*) alloc((size_t)B_ * H_ * 4);
  __bf16* hsA    = (__bf16*)alloc((size_t)NSEQ * H_ * 2);
  __bf16* hsB    = (__bf16*)alloc((size_t)NSEQ * H_ * 2);
  __bf16* predbf = (__bf16*)alloc((size_t)NSEQ * D_ * 2);
  __bf16* dec1bf = (__bf16*)alloc((size_t)NSEQ * H_ * 2);

  float* out_pred   = (float*)d_out;                         // 1984 x 64
  float* out_tgt    = out_pred + (size_t)NSEQ * D_;          // 1984 x 64
  float* out_logits = out_tgt  + (size_t)NSEQ * D_;          // 1984 x 256

  auto gemm = [&](const __bf16* A, const __bf16* Bt, const float* bias,
                  float* C, __bf16* Cbf, int M, int N, int K, int relu) {
    dim3 g(N / 64, M / 64);
    k_gemm_bf16<<<g, 256, 0, stream>>>(A, Bt, bias, C, Cbf, M, N, K, relu);
  };

  // ---- weight prep ----
  k_transpose_to_bf16<<<nb256(IN_F * D_), 256, 0, stream>>>(
      (const float*)d_in[IP_EMB_W], emb_wt, IN_F, D_);
  for (int l = 0; l < 4; ++l) {
    k_transpose_to_bf16<<<nb256(D_ * D_), 256, 0, stream>>>(
        (const float*)d_in[IP_LAYER0 + 5 * l + 4], root_wt + (size_t)l * D_ * D_, D_, D_);
    for (int r = 0; r < R_; ++r)
      k_transpose_to_bf16<<<nb256(D_ * D_), 256, 0, stream>>>(
          (const float*)d_in[IP_LAYER0 + 5 * l + r], conv_wt + (size_t)(l * 3 + r) * D_ * D_, D_, D_);
  }
  k_transpose_to_bf16<<<nb256(H_ * D_), 256, 0, stream>>>(
      (const float*)d_in[IP_HEAD_W], head_wt, H_, D_);
  k_transpose_to_bf16<<<nb256(D_ * H_), 256, 0, stream>>>(
      (const float*)d_in[IP_DEC_W1], d1t, D_, H_);
  k_transpose_to_bf16<<<nb256(H_ * OUT_), 256, 0, stream>>>(
      (const float*)d_in[IP_DEC_W2], d2t, H_, OUT_);
  for (int l = 0; l < 2; ++l) {
    k_f32_to_bf16<<<nb256(4 * H_ * H_), 256, 0, stream>>>(
        (const float*)d_in[IP_LSTM0 + 4 * l + 3], wih[l], (size_t)4 * H_ * H_);
    k_f32_to_bf16<<<nb256(4 * H_ * H_), 256, 0, stream>>>(
        (const float*)d_in[IP_LSTM0 + 4 * l + 2], whh[l], (size_t)4 * H_ * H_);
    k_add_bias<<<nb256(4 * H_), 256, 0, stream>>>(
        (const float*)d_in[IP_LSTM0 + 4 * l + 1], (const float*)d_in[IP_LSTM0 + 4 * l + 0],
        bsum[l], 4 * H_);
  }

  // ---- relational GNN encode (shared weights, run for states then goals) ----
  auto encode = [&](const float* x, const int* edges, int n, int E,
                    float* emb_out, int n_graphs) {
    size_t nf = (size_t)n * D_;
    k_f32_to_bf16<<<nb256((size_t)n * IN_F), 256, 0, stream>>>(x, x_bf, (size_t)n * IN_F);
    gemm(x_bf, emb_wt, (const float*)d_in[IP_EMB_B], h32, h_bf, n, D_, IN_F, 0);
    for (int l = 0; l < 4; ++l) {
      gemm(h_bf, root_wt + (size_t)l * D_ * D_,
           (const float*)d_in[IP_LAYER0 + 5 * l + 3], out32, nullptr, n, D_, D_, 0);
      for (int r = 0; r < R_; ++r) {
        gemm(h_bf, conv_wt + (size_t)(l * 3 + r) * D_ * D_, nullptr, msg32, nullptr,
             n, D_, D_, 0);
        k_fill_u32<<<nb256(nf), 256, 0, stream>>>(agg, MAPPED_NEG_INF, nf);
        const int* src = edges + (size_t)r * 2 * E;
        k_scatter_max<<<nb256((size_t)E * 16), 256, 0, stream>>>(msg32, src, src + E, agg, E);
        k_accum_agg<<<nb256(nf), 256, 0, stream>>>(out32, agg, nf);
      }
      k_relu_store<<<nb256(nf), 256, 0, stream>>>(out32, h32, h_bf, nf);
    }
    k_segment_sum<<<nb256((size_t)n_graphs * D_), 256, 0, stream>>>(h32, emb_out, n_graphs);
  };

  encode((const float*)d_in[IN_X_STATES], (const int*)d_in[IN_EDGE_S], N_S, E_S, emb_s, B_ * T_);
  encode((const float*)d_in[IN_X_GOAL],   (const int*)d_in[IN_EDGE_G], N_G, E_G, emb_g, B_);

  // ---- build LSTM input + tgt output ----
  k_build_input<<<nb256((size_t)NSEQ * D_), 256, 0, stream>>>(emb_s, emb_g, X0bf, out_tgt);

  // ---- LSTM layers (per-timestep WMMA GEMM + pointwise) ----
  const __bf16* layer_in = X0bf;
  __bf16* layer_out[2] = {hsA, hsB};
  for (int l = 0; l < 2; ++l) {
    gemm(layer_in, wih[l], bsum[l], Xpre, nullptr, NSEQ, 4 * H_, 2 * D_, 0);
    k_fill_u32<<<nb256((size_t)B_ * H_ / 2), 256, 0, stream>>>((u32*)hcur, 0u, (size_t)B_ * H_ / 2);
    k_fill_u32<<<nb256((size_t)B_ * H_), 256, 0, stream>>>((u32*)cbuf, 0u, (size_t)B_ * H_);
    for (int t = 0; t < T_ - 1; ++t) {
      gemm(hcur, whh[l], nullptr, gates, nullptr, B_, 4 * H_, H_, 0);
      k_lstm_point<<<nb256((size_t)B_ * H_), 256, 0, stream>>>(gates, Xpre, cbuf, hcur,
                                                               layer_out[l], t);
    }
    layer_in = layer_out[l];
  }

  // ---- heads ----
  gemm(hsB, head_wt, (const float*)d_in[IP_HEAD_B], out_pred, predbf, NSEQ, D_, H_, 0);
  gemm(predbf, d1t, (const float*)d_in[IP_DEC_B1], nullptr, dec1bf, NSEQ, H_, D_, 1);
  gemm(dec1bf, d2t, (const float*)d_in[IP_DEC_B2], out_logits, nullptr, NSEQ, OUT_, H_, 0);
}
